// BiLSTMWithCRF_11562051961097
// MI455X (gfx1250) — compile-verified, matrix operand order checked by reading, not run on hardware
//
#include <hip/hip_runtime.h>
#include <math.h>

// ---- problem constants (from reference) ----
#define BB   64      // batch
#define TT   256     // seq len
#define KK   32      // CRF states
#define HH   512     // hidden
#define EE   300     // embed dim
#define EPAD 320     // embed dim padded to multiple of 32
#define G4H  2048    // 4*H gate width
#define NBLK 8       // workgroups per direction in the recurrence

typedef __attribute__((ext_vector_type(16))) __bf16        v16bf;
typedef __attribute__((ext_vector_type(8)))  float         v8f;
typedef __attribute__((ext_vector_type(8)))  unsigned int  v8u;
typedef __attribute__((ext_vector_type(4)))  unsigned int  u32x4;
typedef __attribute__((ext_vector_type(8)))  int           i32x8;
typedef __attribute__((ext_vector_type(4)))  int           i32x4;

__device__ __forceinline__ float bf2f(unsigned short u) {
    unsigned int x = ((unsigned int)u) << 16;
    return __builtin_bit_cast(float, x);
}
__device__ __forceinline__ unsigned short f2bf(float f) {
    unsigned int x = __builtin_bit_cast(unsigned int, f);
    x += 0x7FFFu + ((x >> 16) & 1u);   // round-to-nearest-even
    return (unsigned short)(x >> 16);
}
// K-offset of VGPR v (pair of bf16) in a 16-bit 16x32 A / 32x16 B fragment
__device__ __forceinline__ int kbase_of(int v, int half) {
    return ((v < 4) ? 0 : 16) + half * 8 + (v & 3) * 2;
}
__device__ __forceinline__ v8f wmma_bf16(v8u a, v8u b, v8f c) {
    v16bf av = __builtin_bit_cast(v16bf, a);
    v16bf bv = __builtin_bit_cast(v16bf, b);
    return __builtin_amdgcn_wmma_f32_16x16x32_bf16(false, av, false, bv,
                                                   (short)0, c, false, false);
}
__device__ __forceinline__ float sigf(float x) { return 1.0f / (1.0f + __expf(-x)); }

// ---- f32 -> bf16 weight conversion with optional column padding ----
__global__ void convert_pad_kernel(const float* __restrict__ src,
                                   unsigned short* __restrict__ dst,
                                   int cols, int pcols) {
    int r = blockIdx.x;
    for (int c = threadIdx.x; c < pcols; c += blockDim.x)
        dst[(size_t)r * pcols + c] = (c < cols) ? f2bf(src[(size_t)r * cols + c])
                                                : (unsigned short)0;
}

// ---- embedding gather: rows are b*T+t, padded to EPAD, bf16 ----
__global__ void embed_gather_kernel(const int* __restrict__ x,
                                    const float* __restrict__ embW,
                                    unsigned short* __restrict__ A0) {
    int row = blockIdx.x;          // b*T + t
    int idx = x[row];
    int e   = threadIdx.x;         // blockDim = 320
    A0[(size_t)row * EPAD + e] =
        (e < EE) ? f2bf(embW[(size_t)idx * EE + e]) : (unsigned short)0;
}

// ---- big parallel GEMM: C[(t*B+b), n] = sum_k A[(b*T+t), k] * W[n, k]  (bf16 in/out, f32 acc)
// grid = (N/64, M/128), block = 256 (8 waves); wave -> one 16x64 output strip
__global__ __launch_bounds__(256) void gemm_bf16_kernel(
    const unsigned short* __restrict__ A,   // M x Kd, row-major bf16
    const unsigned short* __restrict__ W,   // N x Kd, row-major bf16
    unsigned short* __restrict__ C,         // (T*B) x N, bf16, transposed row layout
    int Kd, int N, int kchunks) {
    int lane = threadIdx.x & 31, wave = threadIdx.x >> 5;
    int half = lane >> 4, lrow = lane & 15;
    int mtile = blockIdx.y * 8 + wave;
    int nbase = blockIdx.x * 4;          // in 16-wide tiles
    int m = mtile * 16 + lrow;
    v8f acc[4] = {};
    for (int kc = 0; kc < kchunks; ++kc) {
        v8u a;
#pragma unroll
        for (int v = 0; v < 8; ++v)
            a[v] = *(const unsigned int*)(A + (size_t)m * Kd + kc * 32 + kbase_of(v, half));
#pragma unroll
        for (int nt = 0; nt < 4; ++nt) {
            int n = (nbase + nt) * 16 + lrow;
            v8u b;
#pragma unroll
            for (int v = 0; v < 8; ++v)
                b[v] = *(const unsigned int*)(W + (size_t)n * Kd + kc * 32 + kbase_of(v, half));
            acc[nt] = wmma_bf16(a, b, acc[nt]);
        }
    }
#pragma unroll
    for (int nt = 0; nt < 4; ++nt)
#pragma unroll
        for (int r = 0; r < 8; ++r) {
            int mg = mtile * 16 + half * 8 + r;       // = b*T + t
            int ng = (nbase + nt) * 16 + lrow;
            int orow = (mg & (TT - 1)) * BB + (mg >> 8);   // t*B + b
            C[(size_t)orow * N + ng] = f2bf(acc[nt][r]);
        }
}

// ---- zero the recurrence state: h double-buffers + barrier counters ----
__global__ void init_state_kernel(unsigned short* __restrict__ hbuf,
                                  unsigned int* __restrict__ ctr) {
    size_t n = (size_t)2 * 2 * BB * HH;
    for (size_t i = (size_t)blockIdx.x * blockDim.x + threadIdx.x; i < n;
         i += (size_t)gridDim.x * blockDim.x)
        hbuf[i] = 0;
    if (blockIdx.x == 0 && threadIdx.x < 2) ctr[threadIdx.x] = 0;
}

// ---- BiLSTM recurrence spread over 8 WGPs per direction.
// grid = 16 blocks (dir = blk/8), 256 threads = 8 waves each.
// Per step: TDM broadcasts prev h (64KB, global L2) into LDS, each wave
// computes 2 (batch-tile x hidden-section) units = all 4 gate tiles,
// updates register-resident c, writes its h-slice into the other global
// buffer, then an atomic-counter grid barrier closes the step.
__global__ __launch_bounds__(256) void lstm_multi_kernel(
    const unsigned short* __restrict__ Xp_f, const unsigned short* __restrict__ Xp_b, // (T*B, 4H) bf16
    const unsigned short* __restrict__ Whh_f, const unsigned short* __restrict__ Whh_b, // (4H, H) bf16
    const float* __restrict__ bih_f, const float* __restrict__ bhh_f,
    const float* __restrict__ bih_b, const float* __restrict__ bhh_b,
    unsigned short* __restrict__ hbuf,   // [2 dir][2 buf][BB*HH] bf16
    unsigned int* __restrict__ ctr,      // [2] barrier counters
    unsigned short* __restrict__ Hout) { // (B*T, 2H) bf16; dir selects column half
    __shared__ __align__(128) unsigned short h_s[BB * HH];   // 64 KB prev-h stage
    const int dir = blockIdx.x / NBLK;
    const int blk = blockIdx.x % NBLK;
    const unsigned short* Xp  = dir ? Xp_b  : Xp_f;
    const unsigned short* Whh = dir ? Whh_b : Whh_f;
    const float* bih = dir ? bih_b : bih_f;
    const float* bhh = dir ? bhh_b : bhh_f;
    unsigned short* hb0 = hbuf + (size_t)dir * 2 * BB * HH;
    unsigned int*   cc  = ctr + dir;

    int tid = threadIdx.x, lane = tid & 31, wave = tid >> 5;
    int half = lane >> 4, lrow = lane & 15;
    unsigned int lds_base = (unsigned int)(uintptr_t)(&h_s[0]);

    float c[2][8];
#pragma unroll
    for (int i = 0; i < 2; ++i)
#pragma unroll
        for (int r = 0; r < 8; ++r) c[i][r] = 0.f;

    int p = 0;
    for (int step = 0; step < TT; ++step) {
        int t = dir ? (TT - 1 - step) : step;

        // ---- stage h[p] (global, L2) -> LDS via Tensor Data Mover ----
        if (tid < 32) {
            unsigned long long ga =
                (unsigned long long)(uintptr_t)(hb0 + (size_t)p * BB * HH);
            u32x4 g0;
            g0[0] = 1u;                                       // count=1, user mode
            g0[1] = lds_base;                                 // lds_addr (bytes)
            g0[2] = (unsigned int)(ga & 0xFFFFFFFFu);         // global_addr[31:0]
            g0[3] = (unsigned int)((ga >> 32) & 0x01FFFFFFu)  // global_addr[56:32]
                    | (2u << 30);                             // type=2 (image/D#)
            i32x8 g1;
            g1[0] = (int)0x00010000;   // wg_mask=0, data_size=1 (2 bytes)
            g1[1] = (int)0x80000000;   // tensor_dim0[15:0]=32768 in bits[31:16]
            g1[2] = (int)0x00010000;   // tensor_dim0[31:16]=0, tensor_dim1=1
            g1[3] = (int)0x80000000;   // tensor_dim1 hi=0, tile_dim0=32768
            g1[4] = 1;                 // tile_dim1=1, tile_dim2=0
            g1[5] = 32768;             // tensor_dim0_stride[31:0]
            g1[6] = 0;                 // stride hi / tensor_dim1_stride lo
            g1[7] = 0;
            i32x4 gz4 = {0, 0, 0, 0};
            i32x8 gz8 = {0, 0, 0, 0, 0, 0, 0, 0};
            __builtin_amdgcn_tensor_load_to_lds(g0, g1, gz4, gz4, gz8, 0);
            __builtin_amdgcn_s_wait_tensorcnt(0);
        }
        __syncthreads();

        // ---- compute this block's 16 units (2 per wave) ----
        for (int i = 0; i < 2; ++i) {
            int unit  = wave * 2 + i;
            int mtile = unit & 3;            // 16-row batch tile
            int jl    = unit >> 2;           // local hidden section 0..3
            int j     = blk * 4 + jl;        // global 16-wide hidden section
            int arow  = mtile * 16 + lrow;
            v8f acc[4] = {};
            for (int kc = 0; kc < HH / 32; ++kc) {
                v8u a;
#pragma unroll
                for (int v = 0; v < 8; ++v)
                    a[v] = *(const unsigned int*)(h_s + arow * HH + kc * 32 + kbase_of(v, half));
#pragma unroll
                for (int g = 0; g < 4; ++g) {
                    int n = g * HH + j * 16 + lrow;
                    v8u b;
#pragma unroll
                    for (int v = 0; v < 8; ++v)
                        b[v] = *(const unsigned int*)(Whh + (size_t)n * HH + kc * 32 + kbase_of(v, half));
                    acc[g] = wmma_bf16(a, b, acc[g]);
                }
            }
            float bsum[4];
#pragma unroll
            for (int g = 0; g < 4; ++g) {
                int n = g * HH + j * 16 + lrow;
                bsum[g] = bih[n] + bhh[n];
            }
            int hidx = j * 16 + lrow;
#pragma unroll
            for (int r = 0; r < 8; ++r) {
                int m = mtile * 16 + half * 8 + r;   // batch index
                float gg[4];
#pragma unroll
                for (int g = 0; g < 4; ++g) {
                    int n = g * HH + j * 16 + lrow;
                    gg[g] = acc[g][r] + bf2f(Xp[(size_t)(t * BB + m) * G4H + n]) + bsum[g];
                }
                float i_ = sigf(gg[0]);
                float f_ = sigf(gg[1]);
                float g_ = tanhf(gg[2]);
                float o_ = sigf(gg[3]);
                c[i][r] = f_ * c[i][r] + i_ * g_;
                unsigned short hv = f2bf(o_ * tanhf(c[i][r]));
                hb0[(size_t)(1 - p) * BB * HH + m * HH + hidx] = hv;  // next-step h
                Hout[(size_t)(m * TT + t) * (2 * HH) + dir * HH + hidx] = hv;
            }
        }

        // ---- cross-workgroup barrier for this direction ----
        __threadfence();
        if (tid == 0) {
            __hip_atomic_fetch_add(cc, 1u, __ATOMIC_RELEASE, __HIP_MEMORY_SCOPE_AGENT);
            unsigned int tgt = (unsigned int)(NBLK * (step + 1));
            while (__hip_atomic_load(cc, __ATOMIC_ACQUIRE, __HIP_MEMORY_SCOPE_AGENT) < tgt)
                __builtin_amdgcn_s_sleep(1);
        }
        __syncthreads();
        p ^= 1;
    }
}

// ---- output projection: y = mask * (h2 @ Wout^T + bout), K=32 -> 2 WMMA n-tiles
__global__ __launch_bounds__(256) void outproj_kernel(
    const unsigned short* __restrict__ H2,      // (B*T, 1024) bf16
    const unsigned short* __restrict__ WoutB,   // (32, 1024) bf16
    const float* __restrict__ bout,
    const int* __restrict__ x,
    float* __restrict__ Y) {                    // (B*T, 32) f32, pre-masked
    int lane = threadIdx.x & 31, wave = threadIdx.x >> 5;
    int half = lane >> 4, lrow = lane & 15;
    int mtile = blockIdx.x * 8 + wave;
    int m = mtile * 16 + lrow;
    v8f acc[2] = {};
    for (int kc = 0; kc < 2 * HH / 32; ++kc) {
        v8u a;
#pragma unroll
        for (int v = 0; v < 8; ++v)
            a[v] = *(const unsigned int*)(H2 + (size_t)m * (2 * HH) + kc * 32 + kbase_of(v, half));
#pragma unroll
        for (int nt = 0; nt < 2; ++nt) {
            int n = nt * 16 + lrow;
            v8u b;
#pragma unroll
            for (int v = 0; v < 8; ++v)
                b[v] = *(const unsigned int*)(WoutB + (size_t)n * (2 * HH) + kc * 32 + kbase_of(v, half));
            acc[nt] = wmma_bf16(a, b, acc[nt]);
        }
    }
#pragma unroll
    for (int nt = 0; nt < 2; ++nt)
#pragma unroll
        for (int r = 0; r < 8; ++r) {
            int mg = mtile * 16 + half * 8 + r;       // b*T + t
            int ng = nt * 16 + lrow;
            float val = acc[nt][r] + bout[ng];
            Y[(size_t)mg * KK + ng] = (x[mg] > 0) ? val : 0.f;
        }
}

// ---- CRF: one wave per batch row, lane = state. gold score + forward logsumexp.
__global__ __launch_bounds__(1024) void crf_kernel(
    const float* __restrict__ Y, const int* __restrict__ x,
    const int* __restrict__ y0, const float* __restrict__ trans,
    float* __restrict__ per_b) {
    __shared__ float tr_s[KK * KK];
    int tid = threadIdx.x;
    if (tid < KK * KK) tr_s[tid] = trans[tid];
    __syncthreads();
    int lane = tid & 31, wave = tid >> 5;
    int b = blockIdx.x * 32 + wave;

    float part = 0.f;
    for (int t = lane; t < TT; t += 32) {
        int cur  = y0[b * TT + t];
        int prev = (t == 0) ? 1 /*SOS*/ : y0[b * TT + t - 1];
        float mk = (x[b * TT + t] > 0) ? 1.f : 0.f;
        part += Y[(size_t)(b * TT + t) * KK + cur] + tr_s[cur * KK + prev] * mk;
    }
#pragma unroll
    for (int off = 16; off > 0; off >>= 1) part += __shfl_down(part, off, 32);
    float gold = __shfl(part, 0, 32);

    float alpha = (lane == 1) ? 0.f : -10000.f;
    for (int t = 0; t < TT; ++t) {
        float mx = -3.0e38f;
        for (int k = 0; k < KK; ++k)
            mx = fmaxf(mx, __shfl(alpha, k, 32) + tr_s[lane * KK + k]);
        float s = 0.f;
        for (int k = 0; k < KK; ++k)
            s += __expf(__shfl(alpha, k, 32) + tr_s[lane * KK + k] - mx);
        alpha = mx + __logf(s) + Y[(size_t)(b * TT + t) * KK + lane];
    }
    float mz = alpha;
#pragma unroll
    for (int off = 16; off > 0; off >>= 1) mz = fmaxf(mz, __shfl_xor(mz, off, 32));
    float sz = __expf(alpha - mz);
#pragma unroll
    for (int off = 16; off > 0; off >>= 1) sz += __shfl_xor(sz, off, 32);
    if (lane == 0) per_b[b] = (mz + __logf(sz)) - gold;
}

__global__ void reduce_kernel(const float* __restrict__ per_b, float* __restrict__ out) {
    if (threadIdx.x == 0) {
        float s = 0.f;
        for (int i = 0; i < BB; ++i) s += per_b[i];
        out[0] = s / (float)BB;
    }
}

extern "C" void kernel_launch(void* const* d_in, const int* in_sizes, int n_in,
                              void* d_out, int out_size, void* d_ws, size_t ws_size,
                              hipStream_t stream) {
    const int*   x     = (const int*)d_in[0];
    const int*   y0    = (const int*)d_in[1];
    const float* embW  = (const float*)d_in[2];
    const float* Wih0f = (const float*)d_in[3];
    const float* Whh0f = (const float*)d_in[4];
    const float* bih0f = (const float*)d_in[5];
    const float* bhh0f = (const float*)d_in[6];
    const float* Wih0b = (const float*)d_in[7];
    const float* Whh0b = (const float*)d_in[8];
    const float* bih0b = (const float*)d_in[9];
    const float* bhh0b = (const float*)d_in[10];
    const float* Wih1f = (const float*)d_in[11];
    const float* Whh1f = (const float*)d_in[12];
    const float* bih1f = (const float*)d_in[13];
    const float* bhh1f = (const float*)d_in[14];
    const float* Wih1b = (const float*)d_in[15];
    const float* Whh1b = (const float*)d_in[16];
    const float* bih1b = (const float*)d_in[17];
    const float* bhh1b = (const float*)d_in[18];
    const float* Wout  = (const float*)d_in[19];
    const float* bout  = (const float*)d_in[20];
    const float* trans = (const float*)d_in[21];

    char* ws = (char*)d_ws;
    size_t off = 0;
    auto alloc = [&](size_t bytes) -> void* {
        void* p = ws + off;
        off = (off + bytes + 255) & ~(size_t)255;
        return p;
    };
    const size_t MROWS = (size_t)BB * TT;   // 16384
    unsigned short* A0   = (unsigned short*)alloc(MROWS * EPAD * 2);
    unsigned short* W0f  = (unsigned short*)alloc((size_t)G4H * EPAD * 2);
    unsigned short* W0b  = (unsigned short*)alloc((size_t)G4H * EPAD * 2);
    unsigned short* W1f  = (unsigned short*)alloc((size_t)G4H * 2 * HH * 2);
    unsigned short* W1b  = (unsigned short*)alloc((size_t)G4H * 2 * HH * 2);
    unsigned short* U0f  = (unsigned short*)alloc((size_t)G4H * HH * 2);
    unsigned short* U0b  = (unsigned short*)alloc((size_t)G4H * HH * 2);
    unsigned short* U1f  = (unsigned short*)alloc((size_t)G4H * HH * 2);
    unsigned short* U1b  = (unsigned short*)alloc((size_t)G4H * HH * 2);
    unsigned short* WoB  = (unsigned short*)alloc((size_t)KK * 2 * HH * 2);
    unsigned short* Xf   = (unsigned short*)alloc(MROWS * G4H * 2);
    unsigned short* Xb   = (unsigned short*)alloc(MROWS * G4H * 2);
    unsigned short* H1   = (unsigned short*)alloc(MROWS * 2 * HH * 2);
    unsigned short* H2   = (unsigned short*)alloc(MROWS * 2 * HH * 2);
    float*          Yb   = (float*)alloc(MROWS * KK * 4);
    float*          perb = (float*)alloc(BB * 4);
    unsigned short* HB   = (unsigned short*)alloc((size_t)2 * 2 * BB * HH * 2);
    unsigned int*   CT   = (unsigned int*)alloc(256);

    // weight conversion (f32 -> bf16, pad E to 320)
    convert_pad_kernel<<<G4H, 256, 0, stream>>>(Wih0f, W0f, EE, EPAD);
    convert_pad_kernel<<<G4H, 256, 0, stream>>>(Wih0b, W0b, EE, EPAD);
    convert_pad_kernel<<<G4H, 256, 0, stream>>>(Wih1f, W1f, 2 * HH, 2 * HH);
    convert_pad_kernel<<<G4H, 256, 0, stream>>>(Wih1b, W1b, 2 * HH, 2 * HH);
    convert_pad_kernel<<<G4H, 256, 0, stream>>>(Whh0f, U0f, HH, HH);
    convert_pad_kernel<<<G4H, 256, 0, stream>>>(Whh0b, U0b, HH, HH);
    convert_pad_kernel<<<G4H, 256, 0, stream>>>(Whh1f, U1f, HH, HH);
    convert_pad_kernel<<<G4H, 256, 0, stream>>>(Whh1b, U1b, HH, HH);
    convert_pad_kernel<<<KK, 256, 0, stream>>>(Wout, WoB, 2 * HH, 2 * HH);

    // embedding gather
    embed_gather_kernel<<<(int)MROWS, EPAD, 0, stream>>>(x, embW, A0);

    dim3 gdim(G4H / 64, (int)MROWS / 128);   // (32, 128)
    // layer 0: input projections, recurrence
    gemm_bf16_kernel<<<gdim, 256, 0, stream>>>(A0, W0f, Xf, EPAD, G4H, EPAD / 32);
    gemm_bf16_kernel<<<gdim, 256, 0, stream>>>(A0, W0b, Xb, EPAD, G4H, EPAD / 32);
    init_state_kernel<<<64, 256, 0, stream>>>(HB, CT);
    lstm_multi_kernel<<<2 * NBLK, 256, 0, stream>>>(Xf, Xb, U0f, U0b,
                                                    bih0f, bhh0f, bih0b, bhh0b,
                                                    HB, CT, H1);
    // layer 1
    gemm_bf16_kernel<<<gdim, 256, 0, stream>>>(H1, W1f, Xf, 2 * HH, G4H, 2 * HH / 32);
    gemm_bf16_kernel<<<gdim, 256, 0, stream>>>(H1, W1b, Xb, 2 * HH, G4H, 2 * HH / 32);
    init_state_kernel<<<64, 256, 0, stream>>>(HB, CT);
    lstm_multi_kernel<<<2 * NBLK, 256, 0, stream>>>(Xf, Xb, U1f, U1b,
                                                    bih1f, bhh1f, bih1b, bhh1b,
                                                    HB, CT, H2);
    // projection + mask, CRF, final mean
    outproj_kernel<<<(int)MROWS / 128, 256, 0, stream>>>(H2, WoB, bout, x, Yb);
    crf_kernel<<<BB / 32, 1024, 0, stream>>>(Yb, x, y0, trans, perb);
    reduce_kernel<<<1, 64, 0, stream>>>(perb, (float*)d_out);
    (void)in_sizes; (void)n_in; (void)out_size; (void)ws_size;
}
